// Attention_7567732376073
// MI455X (gfx1250) — matrix operand order repaired and checked by reading, not compiled, not run on hardware
//
#include <hip/hip_runtime.h>
#include <math.h>

// Problem constants (match reference)
#define B_    2
#define S_    2048
#define HID   2048
#define NH    32
#define NKV   8
#define HD    64
#define HALF_ 32
#define NREP  4
#define KVD   (NKV*HD)   // 512

typedef _Float16 f16;
typedef __attribute__((ext_vector_type(16))) _Float16 v16h;
typedef __attribute__((ext_vector_type(8)))  _Float16 f16x8;
typedef __attribute__((ext_vector_type(8)))  float    v8f;

__device__ __forceinline__ f16x8 ldg8(const f16* p) {
  return *reinterpret_cast<const f16x8*>(p);
}

// A-matrix 16x32 f16 layout (ISA 7.12.2): lane half kg holds K in {kg*8+i, 16+kg*8+i}
__device__ __forceinline__ v16h make_v16(f16x8 lo, f16x8 hi) {
  v16h a;
#pragma unroll
  for (int i = 0; i < 8; ++i) { a[i] = lo[i]; a[i + 8] = hi[i]; }
  return a;
}

// ---- cross-lane xor-reduction (max) within 16-lane rows --------------------
#if __has_builtin(__builtin_amdgcn_permlane16)
__device__ __forceinline__ float permx(float v, unsigned s0, unsigned s1) {
  int i = __builtin_amdgcn_permlane16(__builtin_bit_cast(int, v),
                                      __builtin_bit_cast(int, v),
                                      s0, s1, false, false);
  return __builtin_bit_cast(float, i);
}
__device__ __forceinline__ float rowmax16(float x) {
  x = fmaxf(x, permx(x, 0x67452301u, 0xEFCDAB89u)); // xor 1
  x = fmaxf(x, permx(x, 0x54761032u, 0xDCFE98BAu)); // xor 2
  x = fmaxf(x, permx(x, 0x32107654u, 0xBA98FEDCu)); // xor 4
  x = fmaxf(x, permx(x, 0xFEDCBA98u, 0x76543210u)); // xor 8
  return x;
}
#elif __has_builtin(__builtin_amdgcn_ds_swizzle)
__device__ __forceinline__ float rowmax16(float x) {
#pragma unroll
  for (int m = 1; m < 16; m <<= 1) {
    int y = __builtin_amdgcn_ds_swizzle(__builtin_bit_cast(int, x), 0x1F | (m << 10));
    x = fmaxf(x, __builtin_bit_cast(float, y));
  }
  return x;
}
#else
__device__ __forceinline__ float rowmax16(float x) {
  for (int off = 1; off < 16; off <<= 1) x = fmaxf(x, __shfl_xor(x, off, 32));
  return x;
}
#endif

// ---- f32 -> f16 conversion --------------------------------------------------
__global__ void cvt_f32_to_f16(const float* __restrict__ s, f16* __restrict__ d, int n) {
  int i = blockIdx.x * blockDim.x + threadIdx.x;
  int stride = gridDim.x * blockDim.x;
  for (; i < n; i += stride) d[i] = (f16)s[i];
}

// transpose-convert: dst[n][k] = (f16)src[k][n]   (src is KxN)
__global__ void cvt_transpose_f16(const float* __restrict__ src, f16* __restrict__ dst,
                                  int K, int N) {
  int i = blockIdx.x * blockDim.x + threadIdx.x;
  if (i >= K * N) return;
  int n = i / K, k = i % K;            // consecutive threads -> consecutive k -> coalesced store
  dst[i] = (f16)src[(size_t)k * N + n];
}

// ---- WMMA GEMM: D(MxN) = A(MxK,f16) * B(KxN,f16 given as B^T NxK) ----------
// No LDS: A fragments and B^T-row fragments are contiguous 16B global loads.
// 128 threads = 4 waves; each wave owns 32x64 of D; block tile 128x64.
// OUTMODE: 0 = f16 row-major, 1 = f32 row-major, 2 = f16 scattered as V^T
template <int OUTMODE>
__global__ void gemm_wmma(const f16* __restrict__ A, const f16* __restrict__ BT,
                          f16* __restrict__ Dh, float* __restrict__ Df,
                          int M, int N, int K) {
  const int lane = threadIdx.x & 31;
  const int wid  = threadIdx.x >> 5;
  const int r    = lane & 15;
  const int kg   = lane >> 4;
  const int m0   = blockIdx.x * 128 + wid * 32;
  const int n0   = blockIdx.y * 64;

  const f16* a0p = A + (size_t)(m0 + r) * K;
  const f16* a1p = A + (size_t)(m0 + 16 + r) * K;

  v8f acc0[4] = {}, acc1[4] = {};

  for (int k0 = 0; k0 < K; k0 += 32) {
    v16h a0 = make_v16(ldg8(a0p + k0 + kg * 8), ldg8(a0p + k0 + 16 + kg * 8));
    v16h a1 = make_v16(ldg8(a1p + k0 + kg * 8), ldg8(a1p + k0 + 16 + kg * 8));
#pragma unroll
    for (int nt = 0; nt < 4; ++nt) {
      // B fragment: lane half kg holds K in {kg*8+i, 16+kg*8+i} at column n (B^T row)
      const f16* bp = BT + (size_t)(n0 + nt * 16 + r) * K + k0 + kg * 8;
      v16h b = make_v16(ldg8(bp), ldg8(bp + 16));
      acc0[nt] = __builtin_amdgcn_wmma_f32_16x16x32_f16(false, a0, false, b,
                                                        (short)0, acc0[nt], false, false);
      acc1[nt] = __builtin_amdgcn_wmma_f32_16x16x32_f16(false, a1, false, b,
                                                        (short)0, acc1[nt], false, false);
    }
  }

  // C/D layout: lane -> N = lane&15, VGPR v -> M = v + 8*(lane>>4)
  auto writeHalf = [&](v8f (&acc)[4], int mbase) {
#pragma unroll
    for (int nt = 0; nt < 4; ++nt)
#pragma unroll
      for (int v = 0; v < 8; ++v) {
        int gm = mbase + v;
        int gn = n0 + nt * 16 + r;
        if constexpr (OUTMODE == 1) {
          Df[(size_t)gm * N + gn] = acc[nt][v];
        } else if constexpr (OUTMODE == 0) {
          Dh[(size_t)gm * N + gn] = (f16)acc[nt][v];
        } else {
          int b = gm / S_, s = gm % S_;
          int kv = gn / HD, d = gn % HD;
          Dh[(((size_t)b * NKV + kv) * HD + d) * S_ + s] = (f16)acc[nt][v];
        }
      }
  };
  writeHalf(acc0, m0 + kg * 8);
  writeHalf(acc1, m0 + 16 + kg * 8);
}

// ---- RoPE (in place, f16 storage, f32 math) --------------------------------
__global__ void rope_kernel(f16* __restrict__ t, const float* __restrict__ cosv,
                            const float* __restrict__ sinv, int nheads, int total) {
  int i = blockIdx.x * blockDim.x + threadIdx.x;
  if (i >= total) return;
  int j    = i % HALF_;
  int rest = i / HALF_;
  int h    = rest % nheads;
  int bs   = rest / nheads;          // b*S + s
  int s    = bs % S_;
  float c = cosv[s * HALF_ + j], sn = sinv[s * HALF_ + j];
  size_t base = ((size_t)bs * nheads + h) * HD + 2 * j;
  float tr = (float)t[base], ti = (float)t[base + 1];
  t[base]     = (f16)(tr * c - ti * sn);
  t[base + 1] = (f16)(tr * sn + ti * c);
}

// ---- Flash attention: one wave per (b, h, 16-query tile) -------------------
// K row-major (d contiguous) feeds QK^T B-fragments; V^T (key contiguous)
// feeds PV B-fragments; both are pure b128 global loads. P goes through LDS
// once for the C-layout -> A-layout transpose. Row sums come from a P*ones WMMA.
__global__ void attn_kernel(const f16* __restrict__ Q, const f16* __restrict__ Km,
                            const f16* __restrict__ VT, f16* __restrict__ ctx) {
  __shared__ f16 plds[16][32];

  const int lane = threadIdx.x;
  const int r  = lane & 15;
  const int kg = lane >> 4;
  const int nqt = S_ / 16;
  const int qt = blockIdx.x % nqt;
  const int h  = (blockIdx.x / nqt) % NH;
  const int b  = blockIdx.x / (nqt * NH);
  const int kv = h / NREP;
  const int q0 = qt * 16;

  const f16* Qb = Q  + ((size_t)b * S_ * NH + h) * HD;
  const f16* Kb = Km + ((size_t)b * S_ * NKV + kv) * HD;
  const f16* Vt = VT + ((size_t)b * NKV + kv) * HD * S_;   // [d][key]

  v16h aq[2];
  {
    const f16* qrow = Qb + (size_t)(q0 + r) * NH * HD;
#pragma unroll
    for (int t = 0; t < 2; ++t)
      aq[t] = make_v16(ldg8(qrow + t * 32 + kg * 8), ldg8(qrow + t * 32 + 16 + kg * 8));
  }
  v16h ones;
#pragma unroll
  for (int i = 0; i < 16; ++i) ones[i] = (f16)1.0f;

  float mrun[8], lrun[8], corr[8];
#pragma unroll
  for (int v = 0; v < 8; ++v) { mrun[v] = -1e30f; lrun[v] = 0.f; }
  v8f o[4] = {};

  const float SCL = 0.125f * 1.44269504f;   // 1/sqrt(64) * log2(e)
  const int kend = q0 + 15;

  for (int k0 = 0; k0 <= kend; k0 += 32) {
    // scores: two 16-key sub-tiles, K-dim = head_dim (64) = 2 WMMAs each
    v8f sc[2];
#pragma unroll
    for (int n = 0; n < 2; ++n) {
      const f16* krow = Kb + (size_t)(k0 + n * 16 + r) * NKV * HD;
      v8f c = {};
#pragma unroll
      for (int t = 0; t < 2; ++t) {
        v16h bk = make_v16(ldg8(krow + t * 32 + kg * 8),
                           ldg8(krow + t * 32 + 16 + kg * 8));
        c = __builtin_amdgcn_wmma_f32_16x16x32_f16(false, aq[t], false, bk,
                                                   (short)0, c, false, false);
      }
      sc[n] = c;
    }

    // online softmax (exp2 domain); row m = v + 8*kg lives on this 16-lane half
#pragma unroll
    for (int v = 0; v < 8; ++v) {
      const int qrw = q0 + v + 8 * kg;
      float s0 = sc[0][v] * SCL;
      float s1 = sc[1][v] * SCL;
      if (k0 + r      > qrw) s0 = -1e30f;
      if (k0 + 16 + r > qrw) s1 = -1e30f;
      float mnew = fmaxf(mrun[v], rowmax16(fmaxf(s0, s1)));
      corr[v] = exp2f(mrun[v] - mnew);
      mrun[v] = mnew;
      plds[v + 8 * kg][r]      = (f16)exp2f(s0 - mnew);
      plds[v + 8 * kg][16 + r] = (f16)exp2f(s1 - mnew);
    }
    __syncthreads();

    // P as A-fragment: two contiguous 16B LDS loads
    v16h ap = make_v16(*reinterpret_cast<const f16x8*>(&plds[r][kg * 8]),
                       *reinterpret_cast<const f16x8*>(&plds[r][16 + kg * 8]));

    // row sums via P * ones (every lane gets its row's sum)
    v8f zero = {};
    v8f psum = __builtin_amdgcn_wmma_f32_16x16x32_f16(false, ap, false, ones,
                                                      (short)0, zero, false, false);
#pragma unroll
    for (int v = 0; v < 8; ++v) {
      lrun[v] = lrun[v] * corr[v] + psum[v];
#pragma unroll
      for (int nt = 0; nt < 4; ++nt) o[nt][v] = o[nt][v] * corr[v];
    }

    // PV: B-fragments straight from V^T (key-contiguous)
#pragma unroll
    for (int nt = 0; nt < 4; ++nt) {
      const f16* vp = Vt + (size_t)(nt * 16 + r) * S_ + k0 + kg * 8;
      v16h bv = make_v16(ldg8(vp), ldg8(vp + 16));
      o[nt] = __builtin_amdgcn_wmma_f32_16x16x32_f16(false, ap, false, bv,
                                                     (short)0, o[nt], false, false);
    }
    __syncthreads();
  }

  // normalize and write ctx[b, s, h*HD + d]
#pragma unroll
  for (int v = 0; v < 8; ++v) {
    float inv = 1.0f / lrun[v];
    const int qrw = q0 + v + 8 * kg;
    size_t base = ((size_t)b * S_ + qrw) * HID + (size_t)h * HD;
#pragma unroll
    for (int nt = 0; nt < 4; ++nt)
      ctx[base + nt * 16 + r] = (f16)(o[nt][v] * inv);
  }
}

// ---- launcher ---------------------------------------------------------------
extern "C" void kernel_launch(void* const* d_in, const int* in_sizes, int n_in,
                              void* d_out, int out_size, void* d_ws, size_t ws_size,
                              hipStream_t stream) {
  const float* x  = (const float*)d_in[0];
  const float* wq = (const float*)d_in[1];
  const float* wk = (const float*)d_in[2];
  const float* wv = (const float*)d_in[3];
  const float* wo = (const float*)d_in[4];
  const float* fc = (const float*)d_in[5];
  const float* fs = (const float*)d_in[6];
  // d_in[7] = mask: causal mask applied analytically in attn_kernel
  float* out = (float*)d_out;

  char* ws = (char*)d_ws;
  size_t off = 0;
  auto alloc16 = [&](size_t n) -> f16* {
    f16* p = (f16*)(ws + off);
    off += ((n * sizeof(f16) + 255) & ~(size_t)255);
    return p;
  };
  const size_t NX = (size_t)B_ * S_ * HID;
  f16* xh   = alloc16(NX);                       // x, f16 row-major
  f16* wqT  = alloc16((size_t)HID * HID);        // weights transposed (NxK)
  f16* wkT  = alloc16((size_t)HID * KVD);
  f16* wvT  = alloc16((size_t)HID * KVD);
  f16* woT  = alloc16((size_t)HID * HID);
  f16* qh   = alloc16(NX);                       // (b,s,h,d)
  f16* kh   = alloc16((size_t)B_ * S_ * KVD);    // (b,s,kv,d)
  f16* vT   = alloc16((size_t)B_ * S_ * KVD);    // (b,kv,d,key)
  f16* ctxh = alloc16(NX);                       // (b,s,hid)  ~76 MB total

  {
    int n = (int)NX, nb = (n + 255) / 256; if (nb > 8192) nb = 8192;
    cvt_f32_to_f16<<<nb, 256, 0, stream>>>(x, xh, n);
  }
  auto cvtT = [&](const float* s, f16* d, int K, int N) {
    int total = K * N;
    cvt_transpose_f16<<<(total + 255) / 256, 256, 0, stream>>>(s, d, K, N);
  };
  cvtT(wq, wqT, HID, HID);
  cvtT(wk, wkT, HID, KVD);
  cvtT(wv, wvT, HID, KVD);
  cvtT(wo, woT, HID, HID);

  const int M = B_ * S_;  // 4096
  gemm_wmma<0><<<dim3(M / 128, HID / 64), 128, 0, stream>>>(xh, wqT, qh, nullptr, M, HID, HID);
  gemm_wmma<0><<<dim3(M / 128, KVD / 64), 128, 0, stream>>>(xh, wkT, kh, nullptr, M, KVD, HID);
  gemm_wmma<2><<<dim3(M / 128, KVD / 64), 128, 0, stream>>>(xh, wvT, vT, nullptr, M, KVD, HID);

  const int qpairs = B_ * S_ * NH * HALF_;
  const int kpairs = B_ * S_ * NKV * HALF_;
  rope_kernel<<<qpairs / 256, 256, 0, stream>>>(qh, fc, fs, NH,  qpairs);
  rope_kernel<<<kpairs / 256, 256, 0, stream>>>(kh, fc, fs, NKV, kpairs);

  attn_kernel<<<B_ * NH * (S_ / 16), 32, 0, stream>>>(qh, kh, vT, ctxh);

  gemm_wmma<1><<<dim3(M / 128, HID / 64), 128, 0, stream>>>(ctxh, woT, nullptr, out, M, HID, HID);
}
